// SHGRm_56727928045866
// MI455X (gfx1250) — matrix operand order repaired and checked by reading, not compiled, not run on hardware
//
#include <hip/hip_runtime.h>
#include <hip/hip_bf16.h>
#include <math.h>

typedef __attribute__((ext_vector_type(16))) __bf16 v16bf;
typedef __attribute__((ext_vector_type(8)))  __bf16 v8bf;
typedef __attribute__((ext_vector_type(8)))  float  v8f;

#define WAVES 8
#define BLOCK (WAVES * 32)

// Hardware tanh (TRANS op on gfx1250). Prefer the clang builtin when declared;
// otherwise inline asm. The v_nop satisfies the TRANS->consumer hazard (one
// independent op or V_NOP required before the result is used).
__device__ __forceinline__ float fast_tanh(float x) {
#if __has_builtin(__builtin_amdgcn_tanhf)
  return __builtin_amdgcn_tanhf(x);
#else
  float r;
  asm volatile("v_tanh_f32 %0, %1\n\tv_nop" : "=v"(r) : "v"(x));
  return r;
#endif
}

__device__ __forceinline__ v8f wmma_bf16(v16bf a, v16bf b, v8f c) {
  // (neg_a, A, neg_b, B, c_mod, C, reuse_a, reuse_b)
  return __builtin_amdgcn_wmma_f32_16x16x32_bf16(false, a, false, b, (short)0, c,
                                                 false, false);
}

// A-matrix (16x32 bf16) from row-major LDS tile [16][stride]:
// lane<16: M=lane, K = 0..7 and 16..23 ; lane>=16: M=lane-16, K = 8..15 and 24..31
__device__ __forceinline__ v16bf load_a_tile(const __bf16* t, int stride, int lane) {
  const int row = lane & 15, hi = lane >> 4;
  const __bf16* pr = t + row * stride + hi * 8;
  union { v16bf v; v8bf h[2]; } a;
  a.h[0] = *(const v8bf*)(pr);
  a.h[1] = *(const v8bf*)(pr + 16);
  return a.v;
}

// B-matrix (32x16 bf16) = W^T tile from row-major weights [NOUT][K]:
// lane<16: N=lane, K=kbase+0..7,16..23 ; lane>=16: N=lane-16, K=kbase+8..15,24..31
__device__ __forceinline__ v16bf load_b_tile(const __bf16* w, int stride, int nt,
                                             int kbase, int lane) {
  const int n = lane & 15, hi = lane >> 4;
  const __bf16* pr = w + (nt * 16 + n) * stride + kbase + hi * 8;
  union { v16bf v; v8bf h[2]; } b;
  b.h[0] = *(const v8bf*)(pr);
  b.h[1] = *(const v8bf*)(pr + 16);
  return b.v;
}

// C layout: element (M = v + 8*hi, N = lane&15) in VGPR v. Apply bias+act, store bf16.
template<int ACT, int OSTR>
__device__ __forceinline__ void store_c(__bf16* t, int nt, int lane, v8f c,
                                        const float* bias) {
  const int n = lane & 15, hi = lane >> 4;
  const float b = bias[nt * 16 + n];
  #pragma unroll
  for (int v = 0; v < 8; ++v) {
    float f = c[v] + b;
    if (ACT == 1) f = fmaxf(f, 0.0f);
    if (ACT == 2) f = fast_tanh(f);
    t[(hi * 8 + v) * OSTR + nt * 16 + n] = (__bf16)f;
  }
}

// out[16][NOUT] = act(in[16][K] @ W[NOUT][K]^T + bias). in may alias out:
// all A-tile LDS reads are issued before any store (LDS in-order per wave).
template<int K, int NOUT, int ACT, int OSTR, bool ZFILL>
__device__ __forceinline__ void layer(const __bf16* in, __bf16* out,
                                      const __bf16* w, const float* bias, int lane) {
  constexpr int KT = K / 32;
  constexpr int NT = NOUT / 16;
  v16bf a[KT];
  #pragma unroll
  for (int kt = 0; kt < KT; ++kt) a[kt] = load_a_tile(in + kt * 32, K, lane);
  v8f c[NT];
  #pragma unroll
  for (int nt = 0; nt < NT; ++nt) {
    v8f acc = {0.f, 0.f, 0.f, 0.f, 0.f, 0.f, 0.f, 0.f};
    #pragma unroll
    for (int kt = 0; kt < KT; ++kt) {
      v16bf b = load_b_tile(w, K, nt, kt * 32, lane);
      acc = wmma_bf16(a[kt], b, acc);
    }
    c[nt] = acc;
  }
  #pragma unroll
  for (int nt = 0; nt < NT; ++nt) store_c<ACT, OSTR>(out, nt, lane, c[nt], bias);
  if (ZFILL) {  // zero K-padding columns [NOUT, OSTR) for the next padded layer
    const int n = lane & 15, hi = lane >> 4;
    #pragma unroll
    for (int v = 0; v < 8; ++v)
      out[(hi * 8 + v) * OSTR + NOUT + n] = (__bf16)0.0f;
  }
}

__global__ __launch_bounds__(BLOCK) void mlp32_wmma_kernel(
    const float* __restrict__ x,
    const float* __restrict__ Wa1, const float* __restrict__ ba1,
    const float* __restrict__ Wa2, const float* __restrict__ ba2,
    const float* __restrict__ Wa3, const float* __restrict__ ba3,
    const float* __restrict__ Wa4, const float* __restrict__ ba4,
    const float* __restrict__ Wa5, const float* __restrict__ ba5,
    const float* __restrict__ Wb1, const float* __restrict__ bb1,
    const float* __restrict__ Wb2, const float* __restrict__ bb2,
    const float* __restrict__ Wb3, const float* __restrict__ bb3,
    const float* __restrict__ Wb4, const float* __restrict__ bb4,
    const float* __restrict__ Wb5, const float* __restrict__ bb5,
    const float* __restrict__ Wb6, const float* __restrict__ bb6,
    float* __restrict__ out)
{
  // ---- staged bf16 weights (row-major [out][K], K padded to mult. of 32) ----
  __shared__ __align__(16) __bf16 s_wa2[32 * 64];
  __shared__ __align__(16) __bf16 s_wa3[16 * 32];
  __shared__ __align__(16) __bf16 s_wa4[16 * 32];  // 8x16 real, zero-padded
  __shared__ __align__(16) __bf16 s_wb1[64 * 32];  // 64x31 + zero col at p
  __shared__ __align__(16) __bf16 s_wb2[32 * 64];
  __shared__ __align__(16) __bf16 s_wb3[32 * 32];
  __shared__ __align__(16) __bf16 s_wb4[16 * 32];
  __shared__ __align__(16) __bf16 s_wb5[16 * 32];  // 8x16 real, zero-padded
  __shared__ float s_wa1[64], s_ba1[64], s_ba2[32], s_ba3[16], s_ba4[16];
  __shared__ float s_wa5[8], s_ba5[1];
  __shared__ float s_bb1[64], s_bb2[32], s_bb3[32], s_bb4[16], s_bb5[16];
  __shared__ float s_wb6[8], s_bb6[1];
  // ---- per-wave activation tiles ----
  __shared__ __align__(16) __bf16 s_buf[WAVES][16 * 64];
  __shared__ __align__(16) __bf16 s_xt[WAVES][16 * 32];
  __shared__ float s_xp[WAVES][16];

  const int tid  = threadIdx.x;
  const int lane = tid & 31;
  const int wv   = tid >> 5;
  const int p    = blockIdx.y;

  // ================= cooperative weight staging =================
  for (int i = tid; i < 64; i += BLOCK) {
    s_wa1[i] = Wa1[p * 64 + i];
    s_ba1[i] = ba1[p * 64 + i];
    s_bb1[i] = bb1[p * 64 + i];
  }
  for (int i = tid; i < 32; i += BLOCK) {
    s_ba2[i] = ba2[p * 32 + i];
    s_bb2[i] = bb2[p * 32 + i];
    s_bb3[i] = bb3[p * 32 + i];
  }
  for (int i = tid; i < 16; i += BLOCK) {
    s_ba3[i] = ba3[p * 16 + i];
    s_bb4[i] = bb4[p * 16 + i];
    s_ba4[i] = (i < 8) ? ba4[p * 8 + i] : 0.0f;
    s_bb5[i] = (i < 8) ? bb5[p * 8 + i] : 0.0f;
  }
  for (int i = tid; i < 8; i += BLOCK) {
    s_wa5[i] = Wa5[p * 8 + i];
    s_wb6[i] = Wb6[p * 8 + i];
  }
  if (tid == 0) { s_ba5[0] = ba5[p]; s_bb6[0] = bb6[p]; }

  for (int i = tid; i < 32 * 64; i += BLOCK) {
    s_wa2[i] = (__bf16)Wa2[p * 2048 + i];
    s_wb2[i] = (__bf16)Wb2[p * 2048 + i];
  }
  for (int i = tid; i < 32 * 32; i += BLOCK) s_wb3[i] = (__bf16)Wb3[p * 1024 + i];
  for (int i = tid; i < 16 * 32; i += BLOCK) {
    s_wa3[i] = (__bf16)Wa3[p * 512 + i];
    s_wb4[i] = (__bf16)Wb4[p * 512 + i];
    const int o = i >> 5, k = i & 31;
    const bool real = (o < 8) && (k < 16);
    s_wa4[i] = real ? (__bf16)Wa4[p * 128 + o * 16 + k] : (__bf16)0.0f;
    s_wb5[i] = real ? (__bf16)Wb5[p * 128 + o * 16 + k] : (__bf16)0.0f;
  }
  // Wb1: insert a zero input-column at position p (leave-one-out as a masked GEMM)
  for (int i = tid; i < 64 * 32; i += BLOCK) {
    const int o = i >> 5, j = i & 31;
    __bf16 v = (__bf16)0.0f;
    if (j != p) {
      const int jj = (j < p) ? j : j - 1;  // jj in [0,30]
      v = (__bf16)Wb1[p * 1984 + o * 31 + jj];
    }
    s_wb1[i] = v;
  }
  __syncthreads();

  // ================= per-wave 16-row tile =================
  const int row0 = (blockIdx.x * WAVES + wv) * 16;
  __bf16* buf = s_buf[wv];
  __bf16* xt  = s_xt[wv];
  float*  xp  = s_xp[wv];

  // coalesced x tile load: lane = feature column
  #pragma unroll
  for (int r = 0; r < 16; ++r) {
    const float v = x[(size_t)(row0 + r) * 32 + lane];
    xt[r * 32 + lane] = (__bf16)v;
    if (lane == p) xp[r] = v;
  }

  // ---------------- encoder A: 1 -> 64 -> 32 -> 16 -> 8 -> 1 ----------------
  // layer 1 (K=1) is an outer product: VALU into buf[16][64]
  #pragma unroll 4
  for (int i = 0; i < 32; ++i) {
    const int e = lane + 32 * i;
    const int r = e >> 6, o = e & 63;
    buf[r * 64 + o] = (__bf16)fmaxf(xp[r] * s_wa1[o] + s_ba1[o], 0.0f);
  }
  layer<64, 32, 2, 32, false>(buf, buf, s_wa2, s_ba2, lane);  // 4 WMMA, tanh
  layer<32, 16, 2, 32, true >(buf, buf, s_wa3, s_ba3, lane);  // 1 WMMA, tanh, K-pad
  layer<32, 16, 2, 16, false>(buf, buf, s_wa4, s_ba4, lane);  // 1 WMMA (N padded 8->16)
  if (lane < 16) {  // 8 -> 1 dot, interleaved store a_p at column 2p
    float s = s_ba5[0];
    #pragma unroll
    for (int j = 0; j < 8; ++j) s += (float)buf[lane * 16 + j] * s_wa5[j];
    out[(size_t)(row0 + lane) * 64 + 2 * p] = s;
  }

  // ------------- encoder B: 31 -> 64 -> 32 -> 32 -> 16 -> 8 -> 1 -------------
  layer<32, 64, 1, 64, false>(xt,  buf, s_wb1, s_bb1, lane);  // 4 WMMA, relu
  layer<64, 32, 2, 32, false>(buf, buf, s_wb2, s_bb2, lane);  // 4 WMMA, tanh
  layer<32, 32, 2, 32, false>(buf, buf, s_wb3, s_bb3, lane);  // 2 WMMA, tanh
  layer<32, 16, 2, 32, true >(buf, buf, s_wb4, s_bb4, lane);  // 1 WMMA, tanh, K-pad
  layer<32, 16, 0, 16, false>(buf, buf, s_wb5, s_bb5, lane);  // 1 WMMA, linear
  if (lane < 16) {  // 8 -> 1 dot, interleaved store b_p at column 2p+1
    float s = s_bb6[0];
    #pragma unroll
    for (int j = 0; j < 8; ++j) s += (float)buf[lane * 16 + j] * s_wb6[j];
    out[(size_t)(row0 + lane) * 64 + 2 * p + 1] = s;
  }
}

extern "C" void kernel_launch(void* const* d_in, const int* in_sizes, int n_in,
                              void* d_out, int out_size, void* d_ws, size_t ws_size,
                              hipStream_t stream) {
  (void)n_in; (void)out_size; (void)d_ws; (void)ws_size;
  const float* x   = (const float*)d_in[0];
  const float* Wa1 = (const float*)d_in[1];  const float* ba1 = (const float*)d_in[2];
  const float* Wa2 = (const float*)d_in[3];  const float* ba2 = (const float*)d_in[4];
  const float* Wa3 = (const float*)d_in[5];  const float* ba3 = (const float*)d_in[6];
  const float* Wa4 = (const float*)d_in[7];  const float* ba4 = (const float*)d_in[8];
  const float* Wa5 = (const float*)d_in[9];  const float* ba5 = (const float*)d_in[10];
  const float* Wb1 = (const float*)d_in[11]; const float* bb1 = (const float*)d_in[12];
  const float* Wb2 = (const float*)d_in[13]; const float* bb2 = (const float*)d_in[14];
  const float* Wb3 = (const float*)d_in[15]; const float* bb3 = (const float*)d_in[16];
  const float* Wb4 = (const float*)d_in[17]; const float* bb4 = (const float*)d_in[18];
  const float* Wb5 = (const float*)d_in[19]; const float* bb5 = (const float*)d_in[20];
  const float* Wb6 = (const float*)d_in[21]; const float* bb6 = (const float*)d_in[22];

  const int N = in_sizes[0] / 32;              // 65536 rows
  dim3 grid(N / (16 * WAVES), 32);             // (512, 32 features)
  mlp32_wmma_kernel<<<grid, BLOCK, 0, stream>>>(
      x, Wa1, ba1, Wa2, ba2, Wa3, ba3, Wa4, ba4, Wa5, ba5,
      Wb1, bb1, Wb2, bb2, Wb3, bb3, Wb4, bb4, Wb5, bb5, Wb6, bb6,
      (float*)d_out);
}
